// DeepFRI_84361747628358
// MI455X (gfx1250) — matrix operand order repaired and checked by reading, not compiled, not run on hardware
//
#include <hip/hip_runtime.h>
#include <hip/hip_bf16.h>
#include <math.h>

// ---------------------------------------------------------------------------
// Types (plain ext-vectors: union-safe, trivially copyable)
// ---------------------------------------------------------------------------
typedef __bf16   v16bf __attribute__((ext_vector_type(16)));
typedef float    v8f   __attribute__((ext_vector_type(8)));
typedef unsigned u32x4 __attribute__((ext_vector_type(4)));
typedef float    f32x4 __attribute__((ext_vector_type(4)));

#define DFEAT  1280
#define KSTEP  32
#define KTILES (DFEAT / KSTEP)   // 40
#define NTILES (DFEAT / 16)      // 80
#define HID    128               // D/10
#define D3     (3 * DFEAT)       // 3840

union ABFrag { v16bf v; u32x4 u[2]; };

// ---------------------------------------------------------------------------
// zero fill (n4 = number of float4's)
// ---------------------------------------------------------------------------
__global__ __launch_bounds__(256)
void zero_kernel(float* __restrict__ p, long long n4) {
  long long i = (long long)blockIdx.x * 256 + threadIdx.x;
  if (i < n4) ((f32x4*)p)[i] = (f32x4){0.f, 0.f, 0.f, 0.f};
}

// ---------------------------------------------------------------------------
// degrees via fp32 atomics
// ---------------------------------------------------------------------------
__global__ __launch_bounds__(256)
void degree_kernel(const int* __restrict__ ei, float* __restrict__ outdeg,
                   float* __restrict__ indeg, int E) {
  int i = blockIdx.x * 256 + threadIdx.x;
  if (i < E) {
    atomicAdd(&outdeg[ei[i]],     1.0f);
    atomicAdd(&indeg[ei[E + i]],  1.0f);
  }
}

__global__ __launch_bounds__(256)
void isqrt_kernel(const float* __restrict__ outdeg, const float* __restrict__ indeg,
                  float* __restrict__ oisq, float* __restrict__ iisq, int n) {
  int i = blockIdx.x * 256 + threadIdx.x;
  if (i < n) {
    oisq[i] = rsqrtf(fmaxf(outdeg[i], 1.0f));
    iisq[i] = rsqrtf(fmaxf(indeg[i],  1.0f));
  }
}

// ---------------------------------------------------------------------------
// Pack W (row-major f32 [K=1280][N=1280]) into WMMA B-fragment order, bf16.
// Layout: Wb[ ((nt*KTILES + kt)*32 + lane)*16 + 2v + {0,1} ]
//   lane<16 : K = kt*32 + 2v, 2v+1        col N = nt*16 + (lane&15)
//   lane>=16: K = kt*32 + 16 + 2v, ...+1
// Each lane's fragment = 32 contiguous bytes -> coalesced b128 loads in GEMM.
// ---------------------------------------------------------------------------
__global__ __launch_bounds__(256)
void packW_kernel(const float* __restrict__ W, __bf16* __restrict__ Wb) {
  int idx = blockIdx.x * 256 + threadIdx.x;
  const int total = NTILES * KTILES * 256;   // 819200
  if (idx >= total) return;
  int v    = idx & 7;
  int ln   = (idx >> 3) & 31;
  int rest = idx >> 8;
  int kt   = rest % KTILES;
  int nt   = rest / KTILES;
  int n    = nt * 16 + (ln & 15);
  int kb   = kt * 32 + (ln >> 4) * 16 + 2 * v;
  Wb[(size_t)idx * 2]     = (__bf16)W[(size_t)kb       * DFEAT + n];
  Wb[(size_t)idx * 2 + 1] = (__bf16)W[(size_t)(kb + 1) * DFEAT + n];
}

// ---------------------------------------------------------------------------
// Edge scatter: agg[dst,:] += x[src,:] * oisq[src]   (block per edge)
// ---------------------------------------------------------------------------
__global__ __launch_bounds__(256)
void scatter_kernel(const float* __restrict__ x, const int* __restrict__ ei,
                    const float* __restrict__ oisq, float* __restrict__ agg, int E) {
  int e = blockIdx.x;
  int s = ei[e];
  int d = ei[E + e];
  float sc = oisq[s];
  const float* xr = x   + (size_t)s * DFEAT;
  float*       ar = agg + (size_t)d * DFEAT;
  for (int f = threadIdx.x; f < DFEAT; f += 256)
    atomicAdd(&ar[f], xr[f] * sc);
}

// ---------------------------------------------------------------------------
// Dst-norm + f32 -> bf16 (rows >= n write zeros: clean GEMM padding)
// ---------------------------------------------------------------------------
__global__ __launch_bounds__(256)
void scaleconv_kernel(const float* __restrict__ agg, const float* __restrict__ iisq,
                      __bf16* __restrict__ aggb, int n) {
  int row = blockIdx.y;
  int col = blockIdx.x * 256 + threadIdx.x;
  size_t idx = (size_t)row * DFEAT + col;
  float v = (row < n) ? agg[idx] * iisq[row] : 0.0f;
  aggb[idx] = (__bf16)v;
}

// ---------------------------------------------------------------------------
// WMMA GEMM:  C[PM x 1280] = A_bf16[PM x 1280] @ Wb + bias
// block = 8 waves (256 thr); wave tile 32(M) x 64(N); block tile 128 x 128.
// A frags (contiguous K, matches packed B): lane<16 row m0+l, K {k..k+15};
// lane>=16 same rows, K {k+16..k+31} -> two global b128 each.
// ---------------------------------------------------------------------------
__global__ __launch_bounds__(256)
void gemm_bf16_kernel(const __bf16* __restrict__ A, const __bf16* __restrict__ Bp,
                      const float* __restrict__ bias, float* __restrict__ C) {
  const int tid  = threadIdx.x;
  const int lane = tid & 31;
  const int wave = tid >> 5;
  const int wm   = wave & 3;               // 4 waves along M
  const int wn   = wave >> 2;              // 2 waves along N
  const int m0   = blockIdx.y * 128 + wm * 32;
  const int n0   = blockIdx.x * 128 + wn * 64;

  const int l15 = lane & 15;
  const int hi8 = (lane >> 4) * 8;

  const __bf16* pa0 = A + (size_t)(m0 + l15) * DFEAT + (lane >> 4) * 16;
  const __bf16* pa1 = pa0 + (size_t)16 * DFEAT;

  const __bf16* pb[4];
#pragma unroll
  for (int j = 0; j < 4; ++j) {
    int nt = (n0 >> 4) + j;
    pb[j] = Bp + (size_t)nt * KTILES * 512 + (size_t)lane * 16;
  }

  v8f acc[2][4];
#pragma unroll
  for (int i = 0; i < 2; ++i)
#pragma unroll
    for (int j = 0; j < 4; ++j)
      acc[i][j] = (v8f){0.f,0.f,0.f,0.f,0.f,0.f,0.f,0.f};

  for (int kt = 0; kt < KTILES; ++kt) {
    ABFrag a0, a1, bf[4];
    const __bf16* ka0 = pa0 + kt * KSTEP;
    const __bf16* ka1 = pa1 + kt * KSTEP;
    a0.u[0] = *(const u32x4*)(ka0);
    a0.u[1] = *(const u32x4*)(ka0 + 8);
    a1.u[0] = *(const u32x4*)(ka1);
    a1.u[1] = *(const u32x4*)(ka1 + 8);
#pragma unroll
    for (int j = 0; j < 4; ++j) {
      const __bf16* kb = pb[j] + (size_t)kt * 512;
      bf[j].u[0] = *(const u32x4*)(kb);
      bf[j].u[1] = *(const u32x4*)(kb + 8);
    }
#pragma unroll
    for (int j = 0; j < 4; ++j) {
      acc[0][j] = __builtin_amdgcn_wmma_f32_16x16x32_bf16(
          false, a0.v, false, bf[j].v, (short)0, acc[0][j], false, false);
      acc[1][j] = __builtin_amdgcn_wmma_f32_16x16x32_bf16(
          false, a1.v, false, bf[j].v, (short)0, acc[1][j], false, false);
    }
  }

  // C/D layout: VGPR r, lanes 0-15 -> (M=r, N=lane); lanes 16-31 -> (M=8+r, N=lane-16)
  const int rbase = hi8;   // +8 rows for upper half-wave
#pragma unroll
  for (int j = 0; j < 4; ++j) {
    const int col = n0 + j * 16 + l15;
    const float bv = bias[col];
#pragma unroll
    for (int i = 0; i < 2; ++i) {
      const int rowb = m0 + i * 16 + rbase;
#pragma unroll
      for (int r = 0; r < 8; ++r)
        C[(size_t)(rowb + r) * DFEAT + col] = acc[i][j][r] + bv;
    }
  }
}

// ---------------------------------------------------------------------------
// Column sum / n into gr[l*1280 + col], partial sums + atomics
// ---------------------------------------------------------------------------
__global__ __launch_bounds__(256)
void colsum_kernel(const float* __restrict__ f, float* __restrict__ grl,
                   int n, float invn) {
  int col   = blockIdx.x * 256 + threadIdx.x;
  int chunk = (n + gridDim.y - 1) / gridDim.y;
  int r0 = blockIdx.y * chunk;
  int r1 = r0 + chunk; if (r1 > n) r1 = n;
  float s = 0.f;
  for (int r = r0; r < r1; ++r) s += f[(size_t)r * DFEAT + col];
  atomicAdd(&grl[col], s * invn);
}

// ---------------------------------------------------------------------------
// Head: h = relu(relu(gr) @ Wh1 + bh1); out = tanh(h @ Wh2 + bh2)
// ---------------------------------------------------------------------------
__global__ __launch_bounds__(HID)
void head_kernel(const float* __restrict__ gr,
                 const float* __restrict__ Wh1, const float* __restrict__ bh1,
                 const float* __restrict__ Wh2, const float* __restrict__ bh2,
                 float* __restrict__ out) {
  __shared__ float hs[HID];
  int j = threadIdx.x;
  float s = bh1[j];
  for (int i = 0; i < D3; ++i)
    s += fmaxf(gr[i], 0.0f) * Wh1[(size_t)i * HID + j];
  hs[j] = fmaxf(s, 0.0f);
  __syncthreads();
  if (j < 2) {
    float o = bh2[j];
    for (int k = 0; k < HID; ++k) o += hs[k] * Wh2[(size_t)k * 2 + j];
    out[j] = tanhf(o);
  }
}

// ---------------------------------------------------------------------------
// Host-side orchestration
// ---------------------------------------------------------------------------
extern "C" void kernel_launch(void* const* d_in, const int* in_sizes, int n_in,
                              void* d_out, int out_size, void* d_ws, size_t ws_size,
                              hipStream_t stream) {
  const float* x0   = (const float*)d_in[0];
  const int*   ei   = (const int*)  d_in[1];
  const float* W[3] = {(const float*)d_in[3], (const float*)d_in[5], (const float*)d_in[7]};
  const float* b[3] = {(const float*)d_in[4], (const float*)d_in[6], (const float*)d_in[8]};
  const float* Wh1  = (const float*)d_in[9];
  const float* bh1  = (const float*)d_in[10];
  const float* Wh2  = (const float*)d_in[11];
  const float* bh2  = (const float*)d_in[12];
  float* out = (float*)d_out;

  const int n  = in_sizes[0] / DFEAT;     // 10000
  const int E  = in_sizes[1] / 2;         // 60000
  const int PM = ((n + 127) / 128) * 128; // padded rows for the GEMM

  // workspace carve-out
  size_t cur = 0;
  auto alloc = [&](size_t bytes) -> void* {
    cur = (cur + 255) & ~(size_t)255;
    void* p = (char*)d_ws + cur;
    cur += bytes;
    return p;
  };
  float*  deg  = (float*) alloc(sizeof(float) * 2 * n);     // outdeg | indeg
  float*  oisq = (float*) alloc(sizeof(float) * n);
  float*  iisq = (float*) alloc(sizeof(float) * n);
  float*  gr   = (float*) alloc(sizeof(float) * D3);
  __bf16* Wb   = (__bf16*)alloc(sizeof(__bf16) * 3 * (size_t)DFEAT * DFEAT);
  float*  agg  = (float*) alloc(sizeof(float)  * (size_t)PM * DFEAT);
  __bf16* aggb = (__bf16*)alloc(sizeof(__bf16) * (size_t)PM * DFEAT);
  float*  fA   = (float*) alloc(sizeof(float)  * (size_t)PM * DFEAT);
  float*  fB   = (float*) alloc(sizeof(float)  * (size_t)PM * DFEAT);
  (void)ws_size; (void)n_in; (void)out_size;

  float* outdeg = deg;
  float* indeg  = deg + n;

  // 1. zero degrees + gr
  {
    long long n4 = (long long)(2 * n) / 4;
    zero_kernel<<<(unsigned)((n4 + 255) / 256), 256, 0, stream>>>(deg, n4);
    long long g4 = D3 / 4;
    zero_kernel<<<(unsigned)((g4 + 255) / 256), 256, 0, stream>>>(gr, g4);
  }

  // 2. degrees -> isqrt norms
  degree_kernel<<<(E + 255) / 256, 256, 0, stream>>>(ei, outdeg, indeg, E);
  isqrt_kernel<<<(n + 255) / 256, 256, 0, stream>>>(outdeg, indeg, oisq, iisq, n);

  // 3. pack all three weight matrices into WMMA B-fragment order (bf16)
  {
    const int total = NTILES * KTILES * 256;
    for (int l = 0; l < 3; ++l)
      packW_kernel<<<(total + 255) / 256, 256, 0, stream>>>(
          W[l], Wb + (size_t)l * DFEAT * DFEAT);
  }

  // 4. three GCN layers
  const float* xin = x0;
  float* fbuf[3] = {fA, fB, fA};
  for (int l = 0; l < 3; ++l) {
    long long a4 = (long long)n * DFEAT / 4;
    zero_kernel<<<(unsigned)((a4 + 255) / 256), 256, 0, stream>>>(agg, a4);

    scatter_kernel<<<E, 256, 0, stream>>>(xin, ei, oisq, agg, E);

    scaleconv_kernel<<<dim3(DFEAT / 256, PM), 256, 0, stream>>>(agg, iisq, aggb, n);

    gemm_bf16_kernel<<<dim3(DFEAT / 128, PM / 128), 256, 0, stream>>>(
        aggb, Wb + (size_t)l * DFEAT * DFEAT, b[l], fbuf[l]);

    colsum_kernel<<<dim3(DFEAT / 256, 32), 256, 0, stream>>>(
        fbuf[l], gr + (size_t)l * DFEAT, n, 1.0f / (float)n);

    xin = fbuf[l];
  }

  // 5. MLP head -> 2 outputs
  head_kernel<<<1, HID, 0, stream>>>(gr, Wh1, bh1, Wh2, bh2, out);
}
